// YuanBlockSparseTop2MLP_3332894622520
// MI455X (gfx1250) — matrix operand order repaired and checked by reading, not compiled, not run on hardware
//
#include <hip/hip_runtime.h>
#include <hip/hip_bf16.h>
#include <math.h>

// Problem constants (match reference)
#define E_    8
#define H_    2048
#define F_    5632
#define T_    2048
#define KTOP  2
#define TK    (T_ * KTOP)

// Tiling
#define BM 64
#define BN 64
#define BK 32
#define LDK 40   // 32 + 8 bf16 pad -> 80B row pitch (16B aligned, conflict-free b128)

typedef __attribute__((ext_vector_type(16))) __bf16 v16bf;
typedef __attribute__((ext_vector_type(8)))  float  v8f;

// Workspace layout (bytes)
#define WS_OFFSETS 0                         // (E_+1) ints
#define WS_CURSOR  64                        // E_ ints
#define WS_TOKIDX  128                       // TK ints
#define WS_TOKCW   (128 + TK * 4)            // TK floats
#define WS_XBF     33280                     // T_*H_ bf16 (256B aligned)
#define WS_G       (33280 + T_ * H_ * 2)     // TK*F_ bf16

union Pack8 { __bf16 b[8]; uint4 u; };
union Frag  { uint4 u[2]; v16bf v; };

// ---------------------------------------------------------------------------
// Setup: counting-sort (token,slot) pairs by expert; record routing weight.
// ---------------------------------------------------------------------------
__global__ void moe_setup(const int* __restrict__ sel, const float* __restrict__ rw,
                          int* __restrict__ offsets, int* __restrict__ cursor,
                          int* __restrict__ tokidx, float* __restrict__ tokcw) {
  __shared__ int cnt[E_];
  const int t = threadIdx.x;
  if (t < E_) cnt[t] = 0;
  __syncthreads();
  for (int i = t; i < TK; i += blockDim.x) atomicAdd(&cnt[sel[i]], 1);
  __syncthreads();
  if (t == 0) {
    int acc = 0;
    for (int e = 0; e < E_; ++e) { offsets[e] = acc; acc += cnt[e]; }
    offsets[E_] = acc;
  }
  if (t < E_) cursor[t] = 0;
  __syncthreads();
  for (int i = t; i < TK; i += blockDim.x) {
    const int e = sel[i];
    const int pos = offsets[e] + atomicAdd(&cursor[e], 1);
    tokidx[pos] = i >> 1;      // token id (K=2 slots per token)
    tokcw[pos]  = rw[i];
  }
}

// ---------------------------------------------------------------------------
// fp32 -> bf16 activation conversion (done once, reused by all experts)
// ---------------------------------------------------------------------------
__global__ void cvt_x(const float* __restrict__ x, __bf16* __restrict__ xb) {
  for (int i = blockIdx.x * blockDim.x + threadIdx.x; i < T_ * H_;
       i += gridDim.x * blockDim.x)
    xb[i] = (__bf16)x[i];
}

__global__ void zero_out(float* __restrict__ o) {
  for (int i = blockIdx.x * blockDim.x + threadIdx.x; i < T_ * H_;
       i += gridDim.x * blockDim.x)
    o[i] = 0.0f;
}

// ---------------------------------------------------------------------------
// Fragment loads per CDNA5 wave32 WMMA VGPR layouts (CDNA5 ISA),
// arranged so each fragment is exactly two 16-byte LDS reads (ds_load_b128).
// ---------------------------------------------------------------------------
__device__ __forceinline__ v16bf frag_a(const __bf16 (*sAb)[LDK], int rowbase,
                                        int lane) {
  const int m = rowbase + (lane & 15), h = lane >> 4;
  Frag f;
  f.u[0] = *(const uint4*)(const void*)(&sAb[m][h * 8]);
  f.u[1] = *(const uint4*)(const void*)(&sAb[m][16 + h * 8]);
  return f.v;
}

__device__ __forceinline__ v16bf frag_b(const __bf16 (*sBb)[LDK], int colbase,
                                        int lane) {
  const int n = colbase + (lane & 15), h = lane >> 4;
  Frag f;
  f.u[0] = *(const uint4*)(const void*)(&sBb[n][h * 8]);
  f.u[1] = *(const uint4*)(const void*)(&sBb[n][16 + h * 8]);
  return f.v;
}

__device__ __forceinline__ void cvt_store8(__bf16* dst, const float4 a,
                                           const float4 b) {
  Pack8 p;
  p.b[0] = (__bf16)a.x; p.b[1] = (__bf16)a.y;
  p.b[2] = (__bf16)a.z; p.b[3] = (__bf16)a.w;
  p.b[4] = (__bf16)b.x; p.b[5] = (__bf16)b.y;
  p.b[6] = (__bf16)b.z; p.b[7] = (__bf16)b.w;
  *(uint4*)(void*)dst = p.u;
}

__device__ __forceinline__ float fast_silu(float g) {
  // silu(g) = g * sigmoid(g); fast rcp is plenty for an MLP activation
  return g * __builtin_amdgcn_rcpf(1.0f + __expf(-g));
}

// ---------------------------------------------------------------------------
// GEMM1: G[p, :F] = silu(x.W1g^T) * (x.W1u^T) for gathered rows, bf16 out
// grid: (F/BN, TK/BM, E)
// ---------------------------------------------------------------------------
__device__ __forceinline__ void g1_load(
    __bf16 (*dA)[LDK], __bf16 (*dBg)[LDK], __bf16 (*dBu)[LDK],
    const __bf16* aSrc, const float* pgSrc, const float* puSrc,
    int k, int ar, int aq) {
  *(uint4*)(void*)(&dA[ar][aq * 8]) = *(const uint4*)(const void*)(aSrc + k);
  const float4 g0 = *(const float4*)(const void*)(pgSrc + k);
  const float4 g1 = *(const float4*)(const void*)(pgSrc + k + 4);
  const float4 u0 = *(const float4*)(const void*)(puSrc + k);
  const float4 u1 = *(const float4*)(const void*)(puSrc + k + 4);
  cvt_store8(&dBg[ar][aq * 8], g0, g1);
  cvt_store8(&dBu[ar][aq * 8], u0, u1);
  __builtin_prefetch(pgSrc + k + BK, 0, 1);  // keep weight stream ahead
  __builtin_prefetch(puSrc + k + BK, 0, 1);
}

__device__ __forceinline__ void g1_compute(
    const __bf16 (*sAb)[LDK], const __bf16 (*sBgb)[LDK],
    const __bf16 (*sBub)[LDK], int wy, int wx, int lane,
    v8f& accg0, v8f& accg1, v8f& accu0, v8f& accu1) {
  const v16bf a   = frag_a(sAb, wy * 16, lane);
  const v16bf bg0 = frag_b(sBgb, wx * 32, lane);
  const v16bf bg1 = frag_b(sBgb, wx * 32 + 16, lane);
  const v16bf bu0 = frag_b(sBub, wx * 32, lane);
  const v16bf bu1 = frag_b(sBub, wx * 32 + 16, lane);
  accg0 = __builtin_amdgcn_wmma_f32_16x16x32_bf16(false, a, false, bg0,
                                                  (short)0, accg0, false, false);
  accg1 = __builtin_amdgcn_wmma_f32_16x16x32_bf16(false, a, false, bg1,
                                                  (short)0, accg1, false, false);
  accu0 = __builtin_amdgcn_wmma_f32_16x16x32_bf16(false, a, false, bu0,
                                                  (short)0, accu0, false, false);
  accu1 = __builtin_amdgcn_wmma_f32_16x16x32_bf16(false, a, false, bu1,
                                                  (short)0, accu1, false, false);
}

__global__ __launch_bounds__(256) void moe_gemm1(
    const __bf16* __restrict__ Xb, const float* __restrict__ W1,
    const int* __restrict__ offsets, const int* __restrict__ tokidx,
    __bf16* __restrict__ G) {
  const int e = blockIdx.z;
  const int rbeg = offsets[e];
  const int nrows = offsets[e + 1] - rbeg;
  const int row0 = blockIdx.y * BM;
  if (row0 >= nrows) return;
  const int col0 = blockIdx.x * BN;

  __shared__ __align__(16) __bf16 sA[2][BM][LDK];
  __shared__ __align__(16) __bf16 sBg[2][BN][LDK];
  __shared__ __align__(16) __bf16 sBu[2][BN][LDK];
  __shared__ int sTok[BM];

  const int tid = threadIdx.x;
  if (tid < BM) {
    const int r = row0 + tid;
    sTok[tid] = (r < nrows) ? tokidx[rbeg + r] : 0;  // clamp: pad rows compute
  }                                                  // garbage, masked at store
  __syncthreads();

  const int lane = tid & 31;
  const int wave = tid >> 5;
  const int wy = wave & 3;   // 4 row tiles of 16 -> BM=64
  const int wx = wave >> 2;  // 2 col halves of 32 -> BN=64

  v8f accg0 = {}, accg1 = {}, accu0 = {}, accu1 = {};

  // Hoisted per-thread load addresses (invariant over the K loop)
  const int ar = tid >> 2, aq = tid & 3;  // A row / 8-elem chunk; B uses same
  const __bf16* aSrc = Xb + (size_t)sTok[ar] * H_ + aq * 8;
  const float* pgSrc = W1 + (size_t)e * (2 * F_) * H_ +
                       (size_t)(col0 + ar) * H_ + aq * 8;
  const float* puSrc = pgSrc + (size_t)F_ * H_;

  // Software pipeline with static ping/pong buffers (H_/BK even)
  g1_load(sA[0], sBg[0], sBu[0], aSrc, pgSrc, puSrc, 0, ar, aq);
  __syncthreads();

  int k = 0;
  while (k + 2 * BK < H_) {
    g1_load(sA[1], sBg[1], sBu[1], aSrc, pgSrc, puSrc, k + BK, ar, aq);
    g1_compute(sA[0], sBg[0], sBu[0], wy, wx, lane, accg0, accg1, accu0, accu1);
    __syncthreads();
    g1_load(sA[0], sBg[0], sBu[0], aSrc, pgSrc, puSrc, k + 2 * BK, ar, aq);
    g1_compute(sA[1], sBg[1], sBu[1], wy, wx, lane, accg0, accg1, accu0, accu1);
    __syncthreads();
    k += 2 * BK;
  }
  // Tail: steps k (in buf0) and k+BK
  g1_load(sA[1], sBg[1], sBu[1], aSrc, pgSrc, puSrc, k + BK, ar, aq);
  g1_compute(sA[0], sBg[0], sBu[0], wy, wx, lane, accg0, accg1, accu0, accu1);
  __syncthreads();
  g1_compute(sA[1], sBg[1], sBu[1], wy, wx, lane, accg0, accg1, accu0, accu1);

  // Epilogue: fused SiLU(gate) * up -> bf16 G
  const int n = lane & 15, h = lane >> 4;
  const int colb = col0 + wx * 32;
#pragma unroll
  for (int r = 0; r < 8; ++r) {
    const int m = wy * 16 + h * 8 + r;
    const int p = row0 + m;
    if (p < nrows) {
      const size_t grow = (size_t)(rbeg + p) * F_;
      G[grow + colb + n]      = (__bf16)(fast_silu(accg0[r]) * accu0[r]);
      G[grow + colb + 16 + n] = (__bf16)(fast_silu(accg1[r]) * accu1[r]);
    }
  }
}

// ---------------------------------------------------------------------------
// GEMM2: out[token, :H] += cw * (G_e . W2[e]^T)   grid: (H/BN, TK/BM, E)
// ---------------------------------------------------------------------------
__device__ __forceinline__ void g2_load(
    __bf16 (*dA)[LDK], __bf16 (*dB)[LDK],
    const __bf16* aSrc, const float* pwSrc, int k, int ar, int aq) {
  *(uint4*)(void*)(&dA[ar][aq * 8]) = *(const uint4*)(const void*)(aSrc + k);
  const float4 w0 = *(const float4*)(const void*)(pwSrc + k);
  const float4 w1 = *(const float4*)(const void*)(pwSrc + k + 4);
  cvt_store8(&dB[ar][aq * 8], w0, w1);
  __builtin_prefetch(pwSrc + k + BK, 0, 1);
}

__device__ __forceinline__ void g2_compute(
    const __bf16 (*sAb)[LDK], const __bf16 (*sBb)[LDK],
    int wy, int wx, int lane, v8f& acc0, v8f& acc1) {
  const v16bf a  = frag_a(sAb, wy * 16, lane);
  const v16bf b0 = frag_b(sBb, wx * 32, lane);
  const v16bf b1 = frag_b(sBb, wx * 32 + 16, lane);
  acc0 = __builtin_amdgcn_wmma_f32_16x16x32_bf16(false, a, false, b0,
                                                 (short)0, acc0, false, false);
  acc1 = __builtin_amdgcn_wmma_f32_16x16x32_bf16(false, a, false, b1,
                                                 (short)0, acc1, false, false);
}

__global__ __launch_bounds__(256) void moe_gemm2(
    const __bf16* __restrict__ G, const float* __restrict__ W2,
    const int* __restrict__ offsets, const int* __restrict__ tokidx,
    const float* __restrict__ tokcw, float* __restrict__ out) {
  const int e = blockIdx.z;
  const int rbeg = offsets[e];
  const int nrows = offsets[e + 1] - rbeg;
  const int row0 = blockIdx.y * BM;
  if (row0 >= nrows) return;
  const int col0 = blockIdx.x * BN;  // over H

  __shared__ __align__(16) __bf16 sA[2][BM][LDK];
  __shared__ __align__(16) __bf16 sB[2][BN][LDK];
  __shared__ int sTok[BM];
  __shared__ float sCw[BM];

  const int tid = threadIdx.x;
  if (tid < BM) {
    const int r = row0 + tid;
    const bool v = (r < nrows);
    sTok[tid] = v ? tokidx[rbeg + r] : 0;
    sCw[tid]  = v ? tokcw[rbeg + r] : 0.0f;
  }
  __syncthreads();

  const int lane = tid & 31;
  const int wave = tid >> 5;
  const int wy = wave & 3;
  const int wx = wave >> 2;

  v8f acc0 = {}, acc1 = {};

  // Hoisted per-thread addresses; pad rows clamped (masked at epilogue)
  const int ar = tid >> 2, aq = tid & 3;
  const int pr = rbeg + min(row0 + ar, nrows - 1);
  const __bf16* aSrc = G + (size_t)pr * F_ + aq * 8;
  const float* pwSrc = W2 + (size_t)e * H_ * F_ +
                       (size_t)(col0 + ar) * F_ + aq * 8;

  // Software pipeline with static ping/pong buffers (F_/BK even)
  g2_load(sA[0], sB[0], aSrc, pwSrc, 0, ar, aq);
  __syncthreads();

  int k = 0;
  while (k + 2 * BK < F_) {
    g2_load(sA[1], sB[1], aSrc, pwSrc, k + BK, ar, aq);
    g2_compute(sA[0], sB[0], wy, wx, lane, acc0, acc1);
    __syncthreads();
    g2_load(sA[0], sB[0], aSrc, pwSrc, k + 2 * BK, ar, aq);
    g2_compute(sA[1], sB[1], wy, wx, lane, acc0, acc1);
    __syncthreads();
    k += 2 * BK;
  }
  g2_load(sA[1], sB[1], aSrc, pwSrc, k + BK, ar, aq);
  g2_compute(sA[0], sB[0], wy, wx, lane, acc0, acc1);
  __syncthreads();
  g2_compute(sA[1], sB[1], wy, wx, lane, acc0, acc1);

  // Epilogue: scale by routing weight, accumulate into output
  const int n = lane & 15, h = lane >> 4;
  const int colb = col0 + wx * 32;
#pragma unroll
  for (int r = 0; r < 8; ++r) {
    const int m = wy * 16 + h * 8 + r;
    const int p = row0 + m;
    if (p < nrows) {
      const int tok = sTok[m];
      const float cw = sCw[m];
      unsafeAtomicAdd(&out[(size_t)tok * H_ + colb + n], acc0[r] * cw);
      unsafeAtomicAdd(&out[(size_t)tok * H_ + colb + 16 + n], acc1[r] * cw);
    }
  }
}

// ---------------------------------------------------------------------------
extern "C" void kernel_launch(void* const* d_in, const int* in_sizes, int n_in,
                              void* d_out, int out_size, void* d_ws, size_t ws_size,
                              hipStream_t stream) {
  const float* hs  = (const float*)d_in[0];
  const float* rw  = (const float*)d_in[1];
  const int*   sel = (const int*)d_in[2];
  const float* W1  = (const float*)d_in[3];
  const float* W2  = (const float*)d_in[4];
  float* out = (float*)d_out;
  (void)in_sizes; (void)n_in; (void)out_size; (void)ws_size;

  char* ws = (char*)d_ws;
  int*    offsets = (int*)(ws + WS_OFFSETS);
  int*    cursor  = (int*)(ws + WS_CURSOR);
  int*    tokidx  = (int*)(ws + WS_TOKIDX);
  float*  tokcw   = (float*)(ws + WS_TOKCW);
  __bf16* Xb      = (__bf16*)(ws + WS_XBF);
  __bf16* G       = (__bf16*)(ws + WS_G);

  moe_setup<<<1, 256, 0, stream>>>(sel, rw, offsets, cursor, tokidx, tokcw);
  cvt_x<<<2048, 256, 0, stream>>>(hs, Xb);
  zero_out<<<2048, 256, 0, stream>>>(out);
  moe_gemm1<<<dim3(F_ / BN, TK / BM, E_), 256, 0, stream>>>(Xb, W1, offsets,
                                                            tokidx, G);
  moe_gemm2<<<dim3(H_ / BN, TK / BM, E_), 256, 0, stream>>>(G, W2, offsets,
                                                            tokidx, tokcw, out);
}
